// CrossGenderInfoNCELoss_23733989277984
// MI455X (gfx1250) — matrix operand order repaired and checked by reading, not compiled, not run on hardware
//
#include <hip/hip_runtime.h>

// ---------------------------------------------------------------------------
// CrossGenderInfoNCELoss for MI455X (gfx1250, wave32, WMMA)
//
// sims = A @ B^T / T with rows of A,B unit vectors (|sim/T| <= 14.3), so
// logsumexp == log(sum(exp)) with no overflow risk in fp32 (max row sum
// ~8192*e^14.3 ~ 1.3e10). GEMM runs on v_wmma_f32_16x16x32_f16 with a
// two-term f16 split of the fp32 inputs: dot ~= hi*hi + hi*lo + lo*hi
// (3 WMMAs per K-chunk, ~22-bit effective mantissa, fp32 accumulate).
//
// Each wave owns a 32x16 output slab (2 row-tiles) so every 8 KB B-tile
// feeds 24 WMMAs. Next B tile is prefetched with a constant +16KB offset
// (no per-iteration branch). Diagonal masking (FF/MM) is a scalar-uniform
// branch taken at most twice per wave.
// ---------------------------------------------------------------------------

typedef _Float16 v16h __attribute__((ext_vector_type(16)));
typedef _Float16 v8h  __attribute__((ext_vector_type(8)));
typedef float    v8f  __attribute__((ext_vector_type(8)));

#define NROWS 8192
#define DDIM  128
#define INV_T (1.0f / 0.07f)
#define COLS_PER_SLICE 2048
#define NSLICES (NROWS / COLS_PER_SLICE)

__device__ __forceinline__ v8f wmma_f16(v16h a, v16h b, v8f c) {
    return __builtin_amdgcn_wmma_f32_16x16x32_f16(
        /*neg_a=*/false, a, /*neg_b=*/false, b,
        /*c_mod=*/(short)0, c, /*reuse_a=*/false, /*reuse_b=*/false);
}

// A-matrix 16x32 f16 fragment (ISA 7.12.2), rowp = &A[row][0]:
// lanes 0-15 : M=lane,    K = {0..7} U {16..23}
// lanes 16-31: M=lane-16, K = {8..15} U {24..31}
__device__ __forceinline__ v16h load_fragA(const _Float16* __restrict__ rowp,
                                           int kbase, int lane) {
    const int koff = (lane & 16) ? 8 : 0;
    const _Float16* p = rowp + kbase + koff;
    v8h s0 = *(const v8h*)(p);
    v8h s1 = *(const v8h*)(p + 16);
    v16h f;
#pragma unroll
    for (int i = 0; i < 8; ++i) { f[i] = s0[i]; f[8 + i] = s1[i]; }
    return f;
}

// B-matrix 32x16 f16 fragment (same K order as A),
// rowp = &Mat[colTile + (lane&15)][0]:
// lanes 0-15 : N=lane,    K = {0..7} U {16..23}
// lanes 16-31: N=lane-16, K = {8..15} U {24..31}
__device__ __forceinline__ v16h load_fragB(const _Float16* __restrict__ rowp,
                                           int kbase, int lane) {
    const int koff = (lane & 16) ? 8 : 0;
    const _Float16* p = rowp + kbase + koff;
    v8h s0 = *(const v8h*)(p);
    v8h s1 = *(const v8h*)(p + 16);
    v16h f;
#pragma unroll
    for (int i = 0; i < 8; ++i) { f[i] = s0[i]; f[8 + i] = s1[i]; }
    return f;
}

// ---------------------------------------------------------------------------
// fp32 -> (hi, lo) f16 split for both matrices in one launch
// ---------------------------------------------------------------------------
__global__ __launch_bounds__(256) void convert_split(
    const float* __restrict__ F, const float* __restrict__ M,
    _Float16* __restrict__ fh, _Float16* __restrict__ fl,
    _Float16* __restrict__ mh, _Float16* __restrict__ ml, int n) {
    int i = blockIdx.x * blockDim.x + threadIdx.x;
    if (i < n) {
        float x = F[i];
        _Float16 h = (_Float16)x;
        fh[i] = h;
        fl[i] = (_Float16)(x - (float)h);
    } else if (i < 2 * n) {
        int j = i - n;
        float x = M[j];
        _Float16 h = (_Float16)x;
        mh[j] = h;
        ml[j] = (_Float16)(x - (float)h);
    }
}

__global__ __launch_bounds__(256) void zero_f32(float* __restrict__ p, int n) {
    int i = blockIdx.x * blockDim.x + threadIdx.x;
    if (i < n) p[i] = 0.0f;
}

// ---------------------------------------------------------------------------
// rowSum[i] += sum_j exp((A[i] . B[j]) / T) over one 2048-column slice,
// optionally excluding j == i (SKIP_DIAG).
//
// Block = 256 threads = 8 waves: 2 row-pair slots x 4 column phases.
// Block owns 64 rows; each wave owns 32 rows (two 16-row tiles) and sweeps
// its phase's 16-col tiles with stride 64 across the slice.
// ---------------------------------------------------------------------------
template <bool SKIP_DIAG>
__global__ __launch_bounds__(256) void gemm_exp_rowsum(
    const _Float16* __restrict__ Ah, const _Float16* __restrict__ Al,
    const _Float16* __restrict__ Bh, const _Float16* __restrict__ Bl,
    float* __restrict__ rowSum) {
    const int lane = threadIdx.x & 31;
    const int wave = threadIdx.x >> 5;
    const int rt = wave & 1;   // row-pair slot (0..1) -> 32 rows
    const int ct = wave >> 1;  // column phase (0..3)
    const int r0 = blockIdx.x * 64 + rt * 32;  // tile 0 rows r0..r0+15
    const int r1 = r0 + 16;                    // tile 1 rows r1..r1+15
    const int colBase = blockIdx.y * COLS_PER_SLICE;
    const int colEnd = colBase + COLS_PER_SLICE;
    const int nOff = lane & 15;
    const int hOff = (lane & 16) ? 8 : 0;

    // Resident A fragments: 2 row tiles x 4 K-chunks x (hi, lo).
    v16h ahf[2][4], alf[2][4];
#pragma unroll
    for (int t = 0; t < 2; ++t) {
        const size_t ar = (size_t)((t ? r1 : r0) + nOff) * DDIM;
        const _Float16* aH = Ah + ar;
        const _Float16* aL = Al + ar;
#pragma unroll
        for (int kc = 0; kc < 4; ++kc) {
            ahf[t][kc] = load_fragA(aH, kc * 32, lane);
            alf[t][kc] = load_fragA(aL, kc * 32, lane);
        }
    }

    float rsum[2][8] = {{0.f, 0.f, 0.f, 0.f, 0.f, 0.f, 0.f, 0.f},
                        {0.f, 0.f, 0.f, 0.f, 0.f, 0.f, 0.f, 0.f}};

    for (int jt = colBase + ct * 16; jt < colEnd; jt += 64) {
        const size_t br = (size_t)(jt + nOff) * DDIM;
        const _Float16* bH = Bh + br;
        const _Float16* bL = Bl + br;

        // Pull the next tile's rows (+64 rows = constant +16KB) toward the
        // WGP cache; speculative, silently dropped if past the mapping.
        __builtin_prefetch(bH + 64 * DDIM, 0, 3);
        __builtin_prefetch(bL + 64 * DDIM, 0, 3);

        v8f acc0 = {}, acc1 = {};
#pragma unroll
        for (int kc = 0; kc < 4; ++kc) {
            v16h bh = load_fragB(bH, kc * 32, lane);
            v16h bl = load_fragB(bL, kc * 32, lane);
            acc0 = wmma_f16(ahf[0][kc], bh, acc0);  // hi*hi (tile 0)
            acc1 = wmma_f16(ahf[1][kc], bh, acc1);  // hi*hi (tile 1)
            acc0 = wmma_f16(ahf[0][kc], bl, acc0);  // hi*lo
            acc1 = wmma_f16(ahf[1][kc], bl, acc1);
            acc0 = wmma_f16(alf[0][kc], bh, acc0);  // lo*hi
            acc1 = wmma_f16(alf[1][kc], bh, acc1);
        }

#pragma unroll
        for (int v = 0; v < 8; ++v) {
            rsum[0][v] += __expf(acc0[v] * INV_T);
            rsum[1][v] += __expf(acc1[v] * INV_T);
        }

        if (SKIP_DIAG) {
            // Diagonal can only live in the tile whose column base equals the
            // row-tile base: scalar-uniform, taken at most twice per wave.
            if (jt == r0) {
#pragma unroll
                for (int v = 0; v < 8; ++v)
                    if (hOff + v == nOff) rsum[0][v] -= __expf(acc0[v] * INV_T);
            }
            if (jt == r1) {
#pragma unroll
                for (int v = 0; v < 8; ++v)
                    if (hOff + v == nOff) rsum[1][v] -= __expf(acc1[v] * INV_T);
            }
        }
    }

    // Reduce the 16 columns held across each 16-lane half, then one atomic
    // per (row, phase).
#pragma unroll
    for (int t = 0; t < 2; ++t) {
        const int rowT = (t ? r1 : r0) + hOff;
#pragma unroll
        for (int v = 0; v < 8; ++v) {
            float s = rsum[t][v];
            s += __shfl_xor(s, 8, 32);
            s += __shfl_xor(s, 4, 32);
            s += __shfl_xor(s, 2, 32);
            s += __shfl_xor(s, 1, 32);
            if ((lane & 15) == 0) atomicAdd(&rowSum[rowT + v], s);
        }
    }
}

// ---------------------------------------------------------------------------
// loss_dir = mean_i log1p(sum_neg[i] / sum_pos[i])
// out = 1.5 * loss_f2m + 0.5 * loss_m2f
// ---------------------------------------------------------------------------
__global__ __launch_bounds__(256) void finalize_loss(
    const float* __restrict__ sFM, const float* __restrict__ sMF,
    const float* __restrict__ sFF, const float* __restrict__ sMM,
    float* __restrict__ out) {
    float accF = 0.0f, accM = 0.0f;
    for (int i = threadIdx.x; i < NROWS; i += 256) {
        accF += log1pf(sFF[i] / sFM[i]);  // f2m: neg=F.F, pos row-lse of F.M
        accM += log1pf(sMM[i] / sMF[i]);  // m2f: neg=M.M, pos row-lse of M.F
    }
#pragma unroll
    for (int m = 16; m >= 1; m >>= 1) {
        accF += __shfl_xor(accF, m, 32);
        accM += __shfl_xor(accM, m, 32);
    }
    __shared__ float sf[8], sm[8];
    const int wave = threadIdx.x >> 5;
    if ((threadIdx.x & 31) == 0) { sf[wave] = accF; sm[wave] = accM; }
    __syncthreads();
    if (threadIdx.x == 0) {
        float tF = 0.0f, tM = 0.0f;
#pragma unroll
        for (int w = 0; w < 8; ++w) { tF += sf[w]; tM += sm[w]; }
        const float lossF = tF / (float)NROWS;
        const float lossM = tM / (float)NROWS;
        out[0] = 1.5f * lossF + 0.5f * lossM;
    }
}

// ---------------------------------------------------------------------------
extern "C" void kernel_launch(void* const* d_in, const int* in_sizes, int n_in,
                              void* d_out, int out_size, void* d_ws, size_t ws_size,
                              hipStream_t stream) {
    (void)in_sizes; (void)n_in; (void)out_size; (void)ws_size;
    const float* F = (const float*)d_in[0];  // proj_f (8192x128 f32)
    const float* M = (const float*)d_in[1];  // proj_m (8192x128 f32)

    const size_t nElem = (size_t)NROWS * DDIM;          // 1,048,576
    const size_t halfBytes = nElem * sizeof(_Float16);  // 2 MB

    char* ws = (char*)d_ws;
    _Float16* fh = (_Float16*)(ws + 0 * halfBytes);
    _Float16* fl = (_Float16*)(ws + 1 * halfBytes);
    _Float16* mh = (_Float16*)(ws + 2 * halfBytes);
    _Float16* ml = (_Float16*)(ws + 3 * halfBytes);
    float* sums = (float*)(ws + 4 * halfBytes);  // 4 x 8192 floats
    float* sFM = sums + 0 * NROWS;  // row-lse sums of F.M^T (f2m positives)
    float* sMF = sums + 1 * NROWS;  // row-lse sums of M.F^T (m2f positives)
    float* sFF = sums + 2 * NROWS;  // row sums of F.F^T, diag excluded
    float* sMM = sums + 3 * NROWS;  // row sums of M.M^T, diag excluded

    zero_f32<<<(4 * NROWS + 255) / 256, 256, 0, stream>>>(sums, 4 * NROWS);

    convert_split<<<(int)((2 * nElem + 255) / 256), 256, 0, stream>>>(
        F, M, fh, fl, mh, ml, (int)nElem);

    const dim3 gemmGrid(NROWS / 64, NSLICES);
    gemm_exp_rowsum<false><<<gemmGrid, 256, 0, stream>>>(fh, fl, mh, ml, sFM);
    gemm_exp_rowsum<false><<<gemmGrid, 256, 0, stream>>>(mh, ml, fh, fl, sMF);
    gemm_exp_rowsum<true ><<<gemmGrid, 256, 0, stream>>>(fh, fl, fh, fl, sFF);
    gemm_exp_rowsum<true ><<<gemmGrid, 256, 0, stream>>>(mh, ml, mh, ml, sMM);

    finalize_loss<<<1, 256, 0, stream>>>(sFM, sMF, sFF, sMM, (float*)d_out);
}
